// L2SelfAttention_25632364822840
// MI455X (gfx1250) — matrix operand order repaired and checked by reading, not compiled, not run on hardware
//
#include <hip/hip_runtime.h>

typedef __attribute__((ext_vector_type(16))) _Float16 v16h;
typedef __attribute__((ext_vector_type(8)))  float    v8f;
typedef __attribute__((ext_vector_type(4)))  unsigned int u32x4;
typedef __attribute__((ext_vector_type(8)))  int      i32x8;
typedef __attribute__((ext_vector_type(4)))  int      i32x4;

constexpr int NSEQ   = 2048;
constexpr int DMODEL = 384;
constexpr int NHEAD  = 6;
constexpr int HD     = 64;
constexpr int BATCH  = 4;
constexpr float SCALE_ = 0.125f;   // HEAD_DIM^-0.5 = 64^-0.5
constexpr float EPS_   = 1e-8f;

// D = A(16x32 f16) * B(32x16 f16) + C(16x16 f32)
__device__ inline v8f wmma32(v16h a, v16h b, v8f c) {
  return __builtin_amdgcn_wmma_f32_16x16x32_f16(false, a, false, b, (short)0, c,
                                                false, false);
}

// A-fragment (16x32, f16): lane = (ln, sel); elems 0..15 = row[k0+sel*16 .. +15]
// (contiguous k order, matching the B-fragment)
__device__ inline v16h fragA_h(const _Float16* row, int k0, int sel) {
  v16h r;
  const _Float16* p0 = row + k0 + sel * 16;
  const _Float16* p1 = row + k0 + 8 + sel * 16;
#pragma unroll
  for (int i = 0; i < 8; ++i) { r[i] = p0[i]; r[8 + i] = p1[i]; }
  return r;
}
__device__ inline v16h fragA_f(const float* row, int k0, int sel) {
  v16h r;
  const float* p0 = row + k0 + sel * 16;
  const float* p1 = row + k0 + 8 + sel * 16;
#pragma unroll
  for (int i = 0; i < 8; ++i) { r[i] = (_Float16)p0[i]; r[8 + i] = (_Float16)p1[i]; }
  return r;
}
// B-fragment (32x16, f16): lane (ln, sel) holds B[k = sel*16 .. sel*16+15][ln];
// for B[k][n] = SrcRow_n[k], this is 16 contiguous halves of row n.
__device__ inline v16h fragB_h(const _Float16* p) {
  v16h r;
#pragma unroll
  for (int i = 0; i < 16; ++i) r[i] = p[i];
  return r;
}
__device__ inline v16h fragB_f(const float* p) {
  v16h r;
#pragma unroll
  for (int i = 0; i < 16; ++i) r[i] = (_Float16)p[i];
  return r;
}

// Issue a TDM 1-D tile load (nbytes, multiple of 4) global -> LDS.
// D# per CDNA5 ISA ch.8: group0 = {count=1, lds_addr, global_addr[56:0], type=2},
// group1 = {data_size=4B, tensor_dim0=tile_dim0=ndwords, tile_dim1=1, stride0=ndwords},
// groups 2/3 zero (tensor <= 2D).
__device__ inline void tdm_load_1d(const void* gsrc, void* ldst, unsigned nbytes) {
  const unsigned nd = nbytes >> 2;                       // dwords
  const unsigned long long ga = (unsigned long long)gsrc;
  const unsigned ldsoff = (unsigned)(unsigned long long)ldst;  // flat LDS addr low bits
  u32x4 g0;
  g0[0] = 1u;                                            // count=1, user mode
  g0[1] = ldsoff;                                        // lds_addr
  g0[2] = (unsigned)(ga & 0xFFFFFFFFull);                // global_addr[31:0]
  g0[3] = (unsigned)((ga >> 32) & 0x1FFFFFFull) | (2u << 30);  // [56:32] | type=2
  i32x8 g1;
  g1[0] = (int)(2u << 16);                               // data_size=2 (4 bytes)
  g1[1] = (int)((nd & 0xFFFFu) << 16);                   // tensor_dim0 lo16
  g1[2] = (int)((nd >> 16) | (1u << 16));                // tensor_dim0 hi16 | tensor_dim1=1
  g1[3] = (int)((nd & 0xFFFFu) << 16);                   // tile_dim0 = nd
  g1[4] = 1;                                             // tile_dim1=1, tile_dim2=0
  g1[5] = (int)nd;                                       // tensor_dim0_stride lo32
  g1[6] = 0;
  g1[7] = 0;
  i32x4 z4 = {0, 0, 0, 0};
#if defined(__clang_major__) && (__clang_major__ >= 23)
  i32x8 z8 = {0, 0, 0, 0, 0, 0, 0, 0};
  __builtin_amdgcn_tensor_load_to_lds(g0, g1, z4, z4, z8, 0);
#else
  __builtin_amdgcn_tensor_load_to_lds(g0, g1, z4, z4, 0);
#endif
}

// ---------------- Kernel 1: Q/K/V projections (y = x @ W^T + b) ----------------
// grid (512, 3, 3): x = row-tile, y*8+wave = col-tile (24), z = which matrix.
// Writes Qh/Kh as f16 [B,H,N,64]; V transposed as f16 [B,H,64,N].
__global__ __launch_bounds__(256) void qkv_proj_kernel(
    const float* __restrict__ x,
    const float* __restrict__ wq, const float* __restrict__ bq,
    const float* __restrict__ wk, const float* __restrict__ bk,
    const float* __restrict__ wv, const float* __restrict__ bv,
    _Float16* __restrict__ Qh, _Float16* __restrict__ Kh,
    _Float16* __restrict__ Vt) {
  const int mt   = blockIdx.x;
  const int mat  = blockIdx.z;
  const int wave = threadIdx.x >> 5;
  const int lane = threadIdx.x & 31;
  const int ln   = lane & 15;
  const int sel  = lane >> 4;
  const int ct   = blockIdx.y * 8 + wave;
  const int row0 = mt * 16, col0 = ct * 16;

  const float* w  = (mat == 0) ? wq : (mat == 1) ? wk : wv;
  const float* bb = (mat == 0) ? bq : (mat == 1) ? bk : bv;
  const float* xr = x + (size_t)(row0 + ln) * DMODEL;
  const float* wr = w + (size_t)(col0 + ln) * DMODEL;

  v8f acc = {0.f, 0.f, 0.f, 0.f, 0.f, 0.f, 0.f, 0.f};
#pragma unroll
  for (int k0 = 0; k0 < DMODEL; k0 += 32) {
    v16h a = fragA_f(xr, k0, sel);
    v16h b = fragB_f(wr + k0 + sel * 16);
    acc = wmma32(a, b, acc);
  }
  const float badd = bb[col0 + ln];
  const int gcol = col0 + ln;
  const int h = gcol >> 6, dim = gcol & 63;
#pragma unroll
  for (int r = 0; r < 8; ++r) {
    const int grow = row0 + r + sel * 8;
    const int b_ = grow >> 11, n = grow & (NSEQ - 1);
    const int bh = b_ * NHEAD + h;
    const _Float16 val = (_Float16)(acc[r] + badd);
    if (mat == 2) {
      Vt[((size_t)bh * HD + dim) * NSEQ + n] = val;
    } else {
      _Float16* dst = (mat == 0) ? Qh : Kh;
      dst[((size_t)bh * NSEQ + n) * HD + dim] = val;
    }
  }
}

// ---------------- Kernel 2: L2-distance attention ----------------
// grid (128, 24): x = 16-query strip, y = (b*H + h). Block = 8 waves.
// Wave w covers key super-tiles (32 keys) at key0 = w*32 + t*256, t=0..7.
__global__ __launch_bounds__(256) void l2attn_kernel(
    const _Float16* __restrict__ Qh, const _Float16* __restrict__ Kh,
    const _Float16* __restrict__ Vt,
    float* __restrict__ attn_out, float* __restrict__ o_ws) {
  __shared__ _Float16 qs[16 * HD];       // Q strip (staged via TDM)
  __shared__ float q2s[16];
  __shared__ float rowsum[16];
  __shared__ float invs[16];
  __shared__ float oacc[16 * HD];        // fp32 O accumulation
  __shared__ _Float16 pbuf[8 * 16 * 32]; // per-wave P tile bounce (C-layout -> A-frag)

  const int i0  = blockIdx.x * 16;
  const int bh  = blockIdx.y;
  const int tid = threadIdx.x;
  const int wave = tid >> 5, lane = tid & 31, ln = lane & 15, sel = lane >> 4;

  // stage Q strip (16 rows x 64 halves = 2KB contiguous) with the Tensor Data Mover
  if (tid < 32) {
    tdm_load_1d(Qh + ((size_t)bh * NSEQ + i0) * HD, qs, 16 * HD * sizeof(_Float16));
    __builtin_amdgcn_s_wait_tensorcnt((short)0);
  }
#pragma unroll
  for (int u = 0; u < 4; ++u) oacc[u * 256 + tid] = 0.f;
  if (tid < 16) rowsum[tid] = 0.f;
  __syncthreads();  // publishes TDM-written qs to all waves
  if (tid < 16) {   // q^2 per query row
    float s = 0.f;
#pragma unroll
    for (int j = 0; j < HD; ++j) { float v = (float)qs[tid * HD + j]; s += v * v; }
    q2s[tid] = s;
  }
  __syncthreads();

  const _Float16* qrow = qs + ln * HD;
  const v16h aq0 = fragA_h(qrow, 0, sel);
  const v16h aq1 = fragA_h(qrow, 32, sel);
  float q2r[8];
#pragma unroll
  for (int r = 0; r < 8; ++r) q2r[r] = q2s[r + sel * 8];

  const _Float16* kbase = Kh + (size_t)bh * NSEQ * HD;
  const _Float16* vbase = Vt + (size_t)bh * HD * NSEQ;

  // ---- pass A: unnormalized row sums ----
  float rs[8] = {0.f, 0.f, 0.f, 0.f, 0.f, 0.f, 0.f, 0.f};
  for (int t = 0; t < 8; ++t) {
    const int key0 = wave * 32 + t * 256;
    __builtin_prefetch(kbase + (size_t)(key0 + 256 + ln) * HD, 0, 1);
#pragma unroll
    for (int j = 0; j < 2; ++j) {
      const _Float16* krow = kbase + (size_t)(key0 + j * 16 + ln) * HD;
      v16h bk0 = fragB_h(krow + sel * 16);
      v16h bk1 = fragB_h(krow + 32 + sel * 16);
      float k2p = 0.f;
#pragma unroll
      for (int i = 0; i < 16; ++i) {
        float a = (float)bk0[i], b = (float)bk1[i];
        k2p += a * a + b * b;
      }
      const float k2 = k2p + __shfl_xor(k2p, 16, 32);  // pair half-rows
      v8f s = {0.f, 0.f, 0.f, 0.f, 0.f, 0.f, 0.f, 0.f};
      s = wmma32(aq0, bk0, s);
      s = wmma32(aq1, bk1, s);
#pragma unroll
      for (int r = 0; r < 8; ++r)
        rs[r] += __expf((2.f * s[r] - q2r[r] - k2) * SCALE_);
    }
  }
#pragma unroll
  for (int r = 0; r < 8; ++r) {  // reduce across the 16-lane half
    float v = rs[r];
    v += __shfl_xor(v, 1, 32); v += __shfl_xor(v, 2, 32);
    v += __shfl_xor(v, 4, 32); v += __shfl_xor(v, 8, 32);
    rs[r] = v;
  }
  if (ln == 0) {
#pragma unroll
    for (int r = 0; r < 8; ++r) atomicAdd(&rowsum[r + sel * 8], rs[r]);
  }
  __syncthreads();
  if (tid < 16) invs[tid] = 1.f / (rowsum[tid] + EPS_);
  __syncthreads();
  float invr[8];
#pragma unroll
  for (int r = 0; r < 8; ++r) invr[r] = invs[r + sel * 8];

  // ---- pass B: normalize, emit attn, accumulate O = A @ V ----
  v8f oc[4];
#pragma unroll
  for (int c = 0; c < 4; ++c) oc[c] = (v8f){0.f, 0.f, 0.f, 0.f, 0.f, 0.f, 0.f, 0.f};
  _Float16* pb = pbuf + wave * 512;  // [16][32]
  float* arow = attn_out + ((size_t)bh * NSEQ + i0) * NSEQ;

  for (int t = 0; t < 8; ++t) {
    const int key0 = wave * 32 + t * 256;
#pragma unroll
    for (int j = 0; j < 2; ++j) {
      const _Float16* krow = kbase + (size_t)(key0 + j * 16 + ln) * HD;
      v16h bk0 = fragB_h(krow + sel * 16);
      v16h bk1 = fragB_h(krow + 32 + sel * 16);
      float k2p = 0.f;
#pragma unroll
      for (int i = 0; i < 16; ++i) {
        float a = (float)bk0[i], b = (float)bk1[i];
        k2p += a * a + b * b;
      }
      const float k2 = k2p + __shfl_xor(k2p, 16, 32);
      v8f s = {0.f, 0.f, 0.f, 0.f, 0.f, 0.f, 0.f, 0.f};
      s = wmma32(aq0, bk0, s);
      s = wmma32(aq1, bk1, s);
#pragma unroll
      for (int r = 0; r < 8; ++r) {
        float p = __expf((2.f * s[r] - q2r[r] - k2) * SCALE_) * invr[r];
        arow[(size_t)(r + sel * 8) * NSEQ + key0 + j * 16 + ln] = p;
        pb[(r + sel * 8) * 32 + j * 16 + ln] = (_Float16)p;
      }
    }
    __syncthreads();  // C-layout -> A-fragment via LDS (intra-wave ordering)
    v16h ap = fragA_h(pb + ln * 32, 0, sel);
#pragma unroll
    for (int c = 0; c < 4; ++c) {
      const _Float16* vrow = vbase + (size_t)(c * 16 + ln) * NSEQ + key0;
      v16h bv = fragB_h(vrow + sel * 16);
      oc[c] = wmma32(ap, bv, oc[c]);
    }
    __syncthreads();
  }
  // reduce wave-partial O into LDS
#pragma unroll
  for (int c = 0; c < 4; ++c)
#pragma unroll
    for (int r = 0; r < 8; ++r)
      atomicAdd(&oacc[(r + sel * 8) * HD + c * 16 + ln], oc[c][r]);
  __syncthreads();
  const int b_ = bh / NHEAD, h_ = bh % NHEAD;
#pragma unroll
  for (int u = 0; u < 4; ++u) {
    const int idx = u * 256 + tid;
    const int m = idx >> 6, dim = idx & 63;
    o_ws[(size_t)(b_ * NSEQ + i0 + m) * DMODEL + h_ * HD + dim] = oacc[idx];
  }
}

// ---------------- Kernel 3: output projection (out = O @ wo^T + bo) ----------------
__global__ __launch_bounds__(256) void out_proj_kernel(
    const float* __restrict__ o, const float* __restrict__ wo,
    const float* __restrict__ bo, float* __restrict__ out) {
  const int mt = blockIdx.x;
  const int wave = threadIdx.x >> 5, lane = threadIdx.x & 31;
  const int ln = lane & 15, sel = lane >> 4;
  const int ct = blockIdx.y * 8 + wave;
  const int row0 = mt * 16, col0 = ct * 16;
  const float* orow = o + (size_t)(row0 + ln) * DMODEL;
  const float* wr = wo + (size_t)(col0 + ln) * DMODEL;
  v8f acc = {0.f, 0.f, 0.f, 0.f, 0.f, 0.f, 0.f, 0.f};
#pragma unroll
  for (int k0 = 0; k0 < DMODEL; k0 += 32)
    acc = wmma32(fragA_f(orow, k0, sel), fragB_f(wr + k0 + sel * 16), acc);
  const float badd = bo[col0 + ln];
#pragma unroll
  for (int r = 0; r < 8; ++r)
    out[(size_t)(row0 + r + sel * 8) * DMODEL + col0 + ln] = acc[r] + badd;
}

extern "C" void kernel_launch(void* const* d_in, const int* in_sizes, int n_in,
                              void* d_out, int out_size, void* d_ws, size_t ws_size,
                              hipStream_t stream) {
  const float* x  = (const float*)d_in[0];
  const float* wq = (const float*)d_in[1];
  const float* bq = (const float*)d_in[2];
  const float* wk = (const float*)d_in[3];
  const float* bk = (const float*)d_in[4];
  const float* wv = (const float*)d_in[5];
  const float* bv = (const float*)d_in[6];
  const float* wo = (const float*)d_in[7];
  const float* bo = (const float*)d_in[8];

  float* out  = (float*)d_out;
  float* attn = out + (size_t)BATCH * NSEQ * DMODEL;  // tuple: (out, attn)

  const size_t qelems = (size_t)BATCH * NHEAD * NSEQ * HD;  // 3,145,728 halves each
  _Float16* Qh = (_Float16*)d_ws;
  _Float16* Kh = Qh + qelems;
  _Float16* Vt = Kh + qelems;
  float*  o_ws = (float*)(Vt + qelems);  // [B*N, D] fp32

  qkv_proj_kernel<<<dim3(512, 3, 3), 256, 0, stream>>>(x, wq, bq, wk, bk, wv, bv,
                                                       Qh, Kh, Vt);
  l2attn_kernel<<<dim3(128, 24), 256, 0, stream>>>(Qh, Kh, Vt, attn, o_ws);
  out_proj_kernel<<<dim3(512, 3), 256, 0, stream>>>(o_ws, wo, bo, out);
}
